// DualLayerGNNModel_16063177687559
// MI455X (gfx1250) — matrix operand order repaired and checked by reading, not compiled, not run on hardware
//
#include <hip/hip_runtime.h>

typedef __bf16 bf16;
typedef __attribute__((ext_vector_type(8)))  __bf16 v8bf;
typedef __attribute__((ext_vector_type(16))) __bf16 v16bf;
typedef __attribute__((ext_vector_type(8)))  float  v8f;

#define N_LOW   50000
#define F_LOW   215
#define F_PAD   224      // pad K to multiple of 32 for WMMA
#define H_LOW   256
#define E_LOW   800000
#define T_TYPES 8
#define B_GR    50
#define NU_     5000
#define FU_     6
#define HU_     64
#define EU_     20000
#define NUM_STEPS 3

__device__ __forceinline__ bf16 f2bf(float f) {
    union { float f; unsigned u; } x; x.f = f;
    unsigned r = (x.u + 0x7FFFu + ((x.u >> 16) & 1u)) >> 16;
    unsigned short s = (unsigned short)r;
    bf16 out; __builtin_memcpy(&out, &s, 2); return out;
}
__device__ __forceinline__ float bf2f(bf16 b) {
    unsigned short s; __builtin_memcpy(&s, &b, 2);
    union { unsigned u; float f; } x; x.u = ((unsigned)s) << 16; return x.f;
}
__device__ __forceinline__ float sigm(float x) { return 1.f / (1.f + __expf(-x)); }
__device__ __forceinline__ void atomAddF(float* p, float v) { unsafeAtomicAdd(p, v); }

// ---------------------------------------------------------------------------
// bf16 WMMA GEMM:  C[y] = act(A @ BT[y]^T + bias[y])
//   A  : [M, K]      bf16 row-major (K multiple of 32)
//   BT : [Ncol, K]   bf16 row-major (i.e. B transposed, contiguous along K)
//   C  : [M, Ncol]   fp32 or bf16   (Ncol multiple of 64)
// One wave computes a 16(M) x 64(N) tile: the A fragment is loaded once per
// k-step and reused across 4 B fragments / 4 WMMAs (2.5 b128 loads per WMMA,
// 4x fewer streaming-A refetches than a 16x16 tiling).
// 8 waves per block; blockIdx.y = batch index (for the per-edge-type GEMMs).
// ---------------------------------------------------------------------------
__global__ void wmma_gemm_bf16(const bf16* __restrict__ A,
                               const bf16* __restrict__ BT0,
                               void*       __restrict__ C0,
                               const float* __restrict__ bias0,
                               int M, int Ncol, int K,
                               long btStride, long cStride, long biasStride,
                               int act, int outBf16)
{
    const int wave = threadIdx.x >> 5;
    const int lane = threadIdx.x & 31;
    const int tilesN = Ncol >> 6;                       // 64-wide N tiles
    const long tilesTotal = (long)((M + 15) >> 4) * tilesN;
    long tile = (long)blockIdx.x * (blockDim.x >> 5) + wave;
    if (tile >= tilesTotal) return;                     // wave-uniform exit
    const int tn = (int)(tile % tilesN);
    const int tm = (int)(tile / tilesN);

    const bf16* BT = BT0 + (long)blockIdx.y * btStride;

    int arow = tm * 16 + (lane & 15);
    if (arow >= M) arow = M - 1;                        // clamp (store is masked)
    const bf16* Arow = A + (long)arow * K + ((lane >> 4) << 4);              // + kb (0|16)
    const bf16* Brow = BT + (long)(tn * 64 + (lane & 15)) * K + ((lane >> 4) << 4); // + khalf*16

    v8f acc0 = {0.f,0.f,0.f,0.f,0.f,0.f,0.f,0.f};
    v8f acc1 = acc0, acc2 = acc0, acc3 = acc0;
    const long bstep = (long)16 * K;                    // next 16-column group

    for (int k0 = 0; k0 < K; k0 += 32) {
        v8bf lo = *(const v8bf*)(Arow + k0);            // K = kb .. kb+7
        v8bf hi = *(const v8bf*)(Arow + k0 + 8);        // K = kb+8 .. kb+15
        v16bf a;
#pragma unroll
        for (int i = 0; i < 8; ++i) { a[i] = lo[i]; a[8 + i] = hi[i]; }
        v16bf b0 = *(const v16bf*)(Brow + k0);
        v16bf b1 = *(const v16bf*)(Brow + bstep + k0);
        v16bf b2 = *(const v16bf*)(Brow + 2 * bstep + k0);
        v16bf b3 = *(const v16bf*)(Brow + 3 * bstep + k0);
        acc0 = __builtin_amdgcn_wmma_f32_16x16x32_bf16(false, a, false, b0, (short)0, acc0, false, false);
        acc1 = __builtin_amdgcn_wmma_f32_16x16x32_bf16(false, a, false, b1, (short)0, acc1, false, false);
        acc2 = __builtin_amdgcn_wmma_f32_16x16x32_bf16(false, a, false, b2, (short)0, acc2, false, false);
        acc3 = __builtin_amdgcn_wmma_f32_16x16x32_bf16(false, a, false, b3, (short)0, acc3, false, false);
    }

    const int rbase = tm * 16 + ((lane >> 4) << 3);     // lanes 16-31 hold M = v+8
    v8f accs[4] = {acc0, acc1, acc2, acc3};
#pragma unroll
    for (int j = 0; j < 4; ++j) {
        const int col = tn * 64 + j * 16 + (lane & 15);
        const float bv = bias0 ? bias0[(long)blockIdx.y * biasStride + col] : 0.f;
#pragma unroll
        for (int v = 0; v < 8; ++v) {
            const int row = rbase + v;
            if (row < M) {
                float val = accs[j][v] + bv;
                if (act == 1) val = fmaxf(val, 0.f);
                const long idx = (long)blockIdx.y * cStride + (long)row * Ncol + col;
                if (outBf16) ((bf16*)C0)[idx] = f2bf(val);
                else         ((float*)C0)[idx] = val;
            }
        }
    }
}

// ------------------------- conversion / prep kernels -----------------------
__global__ void k_pad_bf16(const float* __restrict__ x, bf16* __restrict__ y,
                           int M, int K, int Kp) {
    long i = (long)blockIdx.x * blockDim.x + threadIdx.x;
    if (i >= (long)M * Kp) return;
    int r = (int)(i / Kp), c = (int)(i % Kp);
    y[i] = f2bf(c < K ? x[(long)r * K + c] : 0.f);
}
__global__ void k_bt_pad(const float* __restrict__ w, bf16* __restrict__ bt,
                         int K, int Ncol, int Kp) {  // w[K,Ncol] -> bt[Ncol,Kp]
    long i = (long)blockIdx.x * blockDim.x + threadIdx.x;
    if (i >= (long)Ncol * Kp) return;
    int n = (int)(i / Kp), k = (int)(i % Kp);
    bt[i] = f2bf(k < K ? w[(long)k * Ncol + n] : 0.f);
}
__global__ void k_bt_batched(const float* __restrict__ w, bf16* __restrict__ bt,
                             long T, long K, long Ncol) { // w[T,K,N] -> bt[T,N,K]
    long i = (long)blockIdx.x * blockDim.x + threadIdx.x;
    if (i >= T * K * Ncol) return;
    long t = i / (K * Ncol), rem = i % (K * Ncol);
    long n = rem / K, k = rem % K;
    bt[i] = f2bf(w[t * K * Ncol + k * Ncol + n]);
}
__global__ void k_f2bf(const float* __restrict__ x, bf16* __restrict__ y, long n) {
    long i = (long)blockIdx.x * blockDim.x + threadIdx.x;
    if (i < n) y[i] = f2bf(x[i]);
}

// ------------------------- lower GNN kernels -------------------------------
__global__ void k_scatter_lower(const bf16* __restrict__ trans,
                                const int* __restrict__ et,
                                const int* __restrict__ src,
                                const int* __restrict__ dst,
                                float* __restrict__ msg, int E) {
    long tid = (long)blockIdx.x * blockDim.x + threadIdx.x;
    long e = tid >> 5;                 // 32 lanes per edge, 8 feats per lane
    if (e >= E) return;
    int lane = (int)(tid & 31);
    int t = et[e], s = src[e], d = dst[e];
    const v8bf vals = *(const v8bf*)(trans + (((long)t * N_LOW + s) << 8) + lane * 8);
    float* out = msg + ((long)d << 8) + lane * 8;
#pragma unroll
    for (int i = 0; i < 8; ++i) atomAddF(out + i, bf2f(vals[i]));
}
__global__ void k_gru(const float* __restrict__ gi, const float* __restrict__ gh,
                      float* __restrict__ h, bf16* __restrict__ hbf, long NH) {
    long i = (long)blockIdx.x * blockDim.x + threadIdx.x;
    if (i >= NH) return;
    long n = i >> 8; int j = (int)(i & 255);
    const float* gin = gi + n * 768;
    const float* ghn = gh + n * 768;
    float r = sigm(gin[j] + ghn[j]);
    float z = sigm(gin[256 + j] + ghn[256 + j]);
    float c = tanhf(gin[512 + j] + r * ghn[512 + j]);
    float hn = (1.f - z) * c + z * h[i];
    h[i] = hn; hbf[i] = f2bf(hn);
}
__global__ void k_pool(const float* __restrict__ h, const int* __restrict__ batch,
                       float* __restrict__ pooled, float* __restrict__ cnt, long NH) {
    long i = (long)blockIdx.x * blockDim.x + threadIdx.x;
    if (i >= NH) return;
    long n = i >> 8; int j = (int)(i & 255);
    int b = batch[n];
    atomAddF(pooled + (long)b * 256 + j, h[i]);
    if (j == 0) atomAddF(cnt + b, 1.f);
}
__global__ void k_pool_div(const float* __restrict__ pooled, const float* __restrict__ cnt,
                           float* __restrict__ emb) {
    int i = blockIdx.x * blockDim.x + threadIdx.x;
    if (i >= B_GR * 256) return;
    emb[i] = pooled[i] / fmaxf(cnt[i >> 8], 1.f);
}
__global__ void k_gate(const float* __restrict__ emb,
                       const float* __restrict__ w1, const float* __restrict__ b1,
                       const float* __restrict__ w2, const float* __restrict__ b2,
                       float* __restrict__ useup) {
    __shared__ float sh[64];
    int b = blockIdx.x, j = threadIdx.x;
    float acc = b1[j];
    for (int k = 0; k < 256; ++k) acc += emb[b * 256 + k] * w1[k * 64 + j];
    sh[j] = fmaxf(acc, 0.f) * w2[j];
    __syncthreads();
    if (j == 0) {
        float s = b2[0];
        for (int k = 0; k < 64; ++k) s += sh[k];
        useup[b] = sigm(s);
    }
}

// ------------------------- upper GNN kernels -------------------------------
__global__ void k_up_proj(const float* __restrict__ x, const float* __restrict__ w,
                          const float* __restrict__ bias, float* __restrict__ hup) {
    long i = (long)blockIdx.x * blockDim.x + threadIdx.x;
    if (i >= (long)NU_ * HU_) return;
    int n = (int)(i >> 6), j = (int)(i & 63);
    float acc = bias[j];
    for (int k = 0; k < FU_; ++k) acc += x[(long)n * FU_ + k] * w[k * HU_ + j];
    hup[i] = fmaxf(acc, 0.f);
}
__global__ void k_up_scatter(const float* __restrict__ hup, const int* __restrict__ src,
                             const int* __restrict__ dst, float* __restrict__ msgU) {
    long tid = (long)blockIdx.x * blockDim.x + threadIdx.x;
    long e = tid >> 6;
    if (e >= EU_) return;
    int j = (int)(tid & 63);
    atomAddF(msgU + ((long)dst[e] << 6) + j, hup[((long)src[e] << 6) + j]);
}
__global__ void k_up_layer(const float* __restrict__ msgU, const float* __restrict__ w,
                           const float* __restrict__ bias, float* __restrict__ hup) {
    long i = (long)blockIdx.x * blockDim.x + threadIdx.x;
    if (i >= (long)NU_ * HU_) return;
    int n = (int)(i >> 6), j = (int)(i & 63);
    float acc = bias[j];
    for (int k = 0; k < HU_; ++k) acc += msgU[((long)n << 6) + k] * w[k * HU_ + j];
    hup[i] += fmaxf(acc, 0.f);
}
__global__ void k_up_mean(const float* __restrict__ hup, float* __restrict__ upemb) {
    long i = (long)blockIdx.x * blockDim.x + threadIdx.x;
    if (i >= (long)NU_ * HU_) return;
    atomAddF(upemb + (i & 63), hup[i] * (1.f / (float)NU_));
}

// ------------------------- classifier --------------------------------------
__global__ void k_clf1(const float* __restrict__ emb, const float* __restrict__ useup,
                       const float* __restrict__ upemb,
                       const float* __restrict__ w1, const float* __restrict__ b1,
                       float* __restrict__ ch) {
    int i = blockIdx.x * blockDim.x + threadIdx.x;   // B*256
    if (i >= B_GR * 256) return;
    int b = i >> 8, j = i & 255;
    float u = useup[b];
    float acc = b1[j];
    for (int k = 0; k < 256; ++k) {
        float e = emb[b * 256 + k];
        float f = e + u * (k < 64 ? upemb[k] : 0.f);
        acc += e * w1[k * 256 + j] + f * w1[(256 + k) * 256 + j];
    }
    ch[i] = fmaxf(acc, 0.f);
}
__global__ void k_clf2(const float* __restrict__ ch, const float* __restrict__ w2,
                       const float* __restrict__ b2, const float* __restrict__ useup,
                       float* __restrict__ out) {
    int b = blockIdx.x * blockDim.x + threadIdx.x;
    if (b >= B_GR) return;
    float acc = b2[0];
    for (int j = 0; j < 256; ++j) acc += ch[b * 256 + j] * w2[j];
    out[b]        = sigm(acc);
    out[B_GR + b] = useup[b];
}

// ===========================================================================
extern "C" void kernel_launch(void* const* d_in, const int* in_sizes, int n_in,
                              void* d_out, int out_size, void* d_ws, size_t ws_size,
                              hipStream_t stream) {
    const float* x_low   = (const float*)d_in[0];
    const int*   eidx    = (const int*)  d_in[1];   // [2,E]
    const int*   etype   = (const int*)  d_in[2];
    const int*   batch   = (const int*)  d_in[3];
    const float* x_up    = (const float*)d_in[5];
    const int*   eidxU   = (const int*)  d_in[6];   // [2,EU]
    const float* proj_w  = (const float*)d_in[7];
    const float* proj_b  = (const float*)d_in[8];
    const float* edge_w  = (const float*)d_in[9];
    const float* edge_b  = (const float*)d_in[10];
    const float* w_ih    = (const float*)d_in[11];
    const float* w_hh    = (const float*)d_in[12];
    const float* b_ih    = (const float*)d_in[13];
    const float* b_hh    = (const float*)d_in[14];
    const float* up_w    = (const float*)d_in[15];
    const float* up_b    = (const float*)d_in[16];
    const float* up_lw   = (const float*)d_in[17];
    const float* up_lb   = (const float*)d_in[18];
    const float* gate_w1 = (const float*)d_in[19];
    const float* gate_b1 = (const float*)d_in[20];
    const float* gate_w2 = (const float*)d_in[21];
    const float* gate_b2 = (const float*)d_in[22];
    const float* clf_w1  = (const float*)d_in[23];
    const float* clf_b1  = (const float*)d_in[24];
    const float* clf_w2  = (const float*)d_in[25];
    const float* clf_b2  = (const float*)d_in[26];
    float* out = (float*)d_out;

    char* ws = (char*)d_ws;
    size_t off = 0;
    auto alloc = [&](size_t bytes) -> void* {
        off = (off + 255) & ~(size_t)255;
        void* p = ws + off;
        off += bytes;
        return p;
    };
    const long NH  = (long)N_LOW * H_LOW;
    float* h       = (float*)alloc(NH * 4);
    bf16*  hbf     = (bf16*) alloc(NH * 2);
    bf16*  xpad    = (bf16*) alloc((long)N_LOW * F_PAD * 2);
    bf16*  projBT  = (bf16*) alloc((long)H_LOW * F_PAD * 2);
    bf16*  edgeBT  = (bf16*) alloc((long)T_TYPES * H_LOW * H_LOW * 2);
    bf16*  wihBT   = (bf16*) alloc((long)3 * H_LOW * H_LOW * 2);
    bf16*  whhBT   = (bf16*) alloc((long)3 * H_LOW * H_LOW * 2);
    bf16*  trans   = (bf16*) alloc((long)T_TYPES * NH * 2);
    float* msg     = (float*)alloc(NH * 4);
    bf16*  msgbf   = (bf16*) alloc(NH * 2);
    float* gi      = (float*)alloc(NH * 3 * 4);
    float* gh      = (float*)alloc(NH * 3 * 4);
    float* pooled  = (float*)alloc((long)B_GR * H_LOW * 4);
    float* cnt     = (float*)alloc(B_GR * 4);
    float* emb     = (float*)alloc((long)B_GR * H_LOW * 4);
    float* useup   = (float*)alloc(B_GR * 4);
    float* hup     = (float*)alloc((long)NU_ * HU_ * 4);
    float* msgU    = (float*)alloc((long)NU_ * HU_ * 4);
    float* upemb   = (float*)alloc(HU_ * 4);
    float* chbuf   = (float*)alloc((long)B_GR * H_LOW * 4);

    const int TPB = 256;
    auto blocks = [](long n, int tpb) { return (int)((n + tpb - 1) / tpb); };

    // ---- prep: bf16 conversions / transposes ----
    k_pad_bf16<<<blocks((long)N_LOW * F_PAD, TPB), TPB, 0, stream>>>(x_low, xpad, N_LOW, F_LOW, F_PAD);
    k_bt_pad  <<<blocks((long)H_LOW * F_PAD, TPB), TPB, 0, stream>>>(proj_w, projBT, F_LOW, H_LOW, F_PAD);
    k_bt_batched<<<blocks((long)T_TYPES * H_LOW * H_LOW, TPB), TPB, 0, stream>>>(edge_w, edgeBT, T_TYPES, H_LOW, H_LOW);
    k_f2bf<<<blocks((long)3 * H_LOW * H_LOW, TPB), TPB, 0, stream>>>(w_ih, wihBT, (long)3 * H_LOW * H_LOW);
    k_f2bf<<<blocks((long)3 * H_LOW * H_LOW, TPB), TPB, 0, stream>>>(w_hh, whhBT, (long)3 * H_LOW * H_LOW);

    // ---- projection: h = relu(x @ proj_w + proj_b) ----
    {
        long tiles = (long)((N_LOW + 15) / 16) * (H_LOW / 64);
        wmma_gemm_bf16<<<dim3(blocks(tiles, 8)), TPB, 0, stream>>>(
            xpad, projBT, h, proj_b, N_LOW, H_LOW, F_PAD, 0, 0, 0, /*relu*/1, /*f32*/0);
    }
    k_f2bf<<<blocks(NH, TPB), TPB, 0, stream>>>(h, hbf, NH);

    const int* src = eidx;           // edge_index[0]
    const int* dst = eidx + E_LOW;   // edge_index[1]

    for (int step = 0; step < NUM_STEPS; ++step) {
        // trans[t] = h @ edge_w[t] + edge_b[t]  (bf16 out, batched over t)
        {
            long tiles = (long)(N_LOW / 16) * (H_LOW / 64);
            wmma_gemm_bf16<<<dim3(blocks(tiles, 8), T_TYPES), TPB, 0, stream>>>(
                hbf, edgeBT, trans, edge_b, N_LOW, H_LOW, H_LOW,
                (long)H_LOW * H_LOW, NH, H_LOW, /*act*/0, /*bf16*/1);
        }
        // msg = segment_sum(trans[etype, src], dst)
        hipMemsetAsync(msg, 0, NH * 4, stream);
        k_scatter_lower<<<blocks((long)E_LOW * 32, TPB), TPB, 0, stream>>>(trans, etype, src, dst, msg, E_LOW);
        k_f2bf<<<blocks(NH, TPB), TPB, 0, stream>>>(msg, msgbf, NH);
        // gi = msg @ w_ih.T + b_ih ;  gh = h @ w_hh.T + b_hh   (BT == w_ih / w_hh)
        {
            long tiles = (long)(N_LOW / 16) * (3 * H_LOW / 64);
            wmma_gemm_bf16<<<dim3(blocks(tiles, 8)), TPB, 0, stream>>>(
                msgbf, wihBT, gi, b_ih, N_LOW, 3 * H_LOW, H_LOW, 0, 0, 0, 0, 0);
            wmma_gemm_bf16<<<dim3(blocks(tiles, 8)), TPB, 0, stream>>>(
                hbf, whhBT, gh, b_hh, N_LOW, 3 * H_LOW, H_LOW, 0, 0, 0, 0, 0);
        }
        k_gru<<<blocks(NH, TPB), TPB, 0, stream>>>(gi, gh, h, hbf, NH);
    }

    // ---- mean pool per graph + gate ----
    hipMemsetAsync(pooled, 0, (long)B_GR * H_LOW * 4, stream);
    hipMemsetAsync(cnt, 0, B_GR * 4, stream);
    k_pool<<<blocks(NH, TPB), TPB, 0, stream>>>(h, batch, pooled, cnt, NH);
    k_pool_div<<<blocks((long)B_GR * H_LOW, TPB), TPB, 0, stream>>>(pooled, cnt, emb);
    k_gate<<<B_GR, 64, 0, stream>>>(emb, gate_w1, gate_b1, gate_w2, gate_b2, useup);

    // ---- upper GNN ----
    k_up_proj<<<blocks((long)NU_ * HU_, TPB), TPB, 0, stream>>>(x_up, up_w, up_b, hup);
    const int* srcU = eidxU;
    const int* dstU = eidxU + EU_;
    for (int l = 0; l < 2; ++l) {
        hipMemsetAsync(msgU, 0, (long)NU_ * HU_ * 4, stream);
        k_up_scatter<<<blocks((long)EU_ * HU_, TPB), TPB, 0, stream>>>(hup, srcU, dstU, msgU);
        k_up_layer<<<blocks((long)NU_ * HU_, TPB), TPB, 0, stream>>>(
            msgU, up_lw + (long)l * HU_ * HU_, up_lb + (long)l * HU_, hup);
    }
    hipMemsetAsync(upemb, 0, HU_ * 4, stream);
    k_up_mean<<<blocks((long)NU_ * HU_, TPB), TPB, 0, stream>>>(hup, upemb);

    // ---- classifier ----
    k_clf1<<<blocks((long)B_GR * H_LOW, TPB), TPB, 0, stream>>>(emb, useup, upemb, clf_w1, clf_b1, chbuf);
    k_clf2<<<1, 64, 0, stream>>>(chbuf, clf_w2, clf_b2, useup, out);
}
